// HPNLearner_41463614275594
// MI455X (gfx1250) — matrix operand order repaired and checked by reading, not compiled, not run on hardware
//
#include <hip/hip_runtime.h>
#include <hip/hip_bf16.h>

typedef __attribute__((ext_vector_type(16))) _Float16 v16h;
typedef __attribute__((ext_vector_type(8)))  float    v8f;

// ---------------------------------------------------------------------------
// Pack kernel.
// K layout: [branch1 K padded to 32-mult][branch2 K padded to 32-mult].
//  (1) weights into WMMA A-fragment order [kchunk][mt][lane][16] f16
//  (2) per-kv table: {lane-independent offset, ((dy-pad+256)<<16)|(dx-pad+256)}
//      padding entries = {0,0} -> dy=-256 always fails the bounds test
//  (3) bias sum.
// ---------------------------------------------------------------------------
__global__ void pack_wb(const float* __restrict__ w1, const float* __restrict__ b1,
                        const float* __restrict__ w2, const float* __restrict__ b2,
                        _Float16* __restrict__ wpf, int2* __restrict__ tbl,
                        float* __restrict__ bs,
                        int co, int cikk, int K1_32, int K2_32, int k,
                        int st_c, int st_i, int st_j, int wb, int total_wpf)
{
    const int idx = blockIdx.x * blockDim.x + threadIdx.x;
    if (idx < co) bs[idx] = b1[idx] + (w2 ? b2[idx] : 0.0f);

    if (idx < K1_32 + K2_32) {
        int2 e; e.x = 0; e.y = 0;
        int kb, br;
        if (idx < K1_32) { br = 0; kb = idx; }
        else             { br = 1; kb = idx - K1_32; }
        if (kb < cikk) {
            const int dx  = kb % k;
            const int rem = kb / k;
            const int dy  = rem % k;
            const int cin = rem / k;
            const int pad = k >> 1;
            const int dyp = dy - pad, dxp = dx - pad;
            e.x = br ? (cin * st_c + dyp * wb + dxp)
                     : (cin * st_c + dyp * st_i + dxp * st_j);
            e.y = ((dyp + 256) << 16) | (dxp + 256);
        }
        tbl[idx] = e;
    }

    if (idx >= total_wpf) return;
    const int t    = idx & 15;          // half within lane
    const int l    = (idx >> 4) & 31;   // lane
    const int tile = idx >> 9;          // (kc, mt)
    const int mt_n = co >> 4;
    const int mt   = tile % mt_n;
    const int kc   = tile / mt_n;
    const int v    = t >> 1;
    const int kv   = kc * 32 + ((v >= 4) ? 16 : 0) + ((l >= 16) ? 8 : 0)
                   + (v & 3) * 2 + (t & 1);
    const int oc   = mt * 16 + (l & 15);
    float val = 0.0f;
    if (kv < K1_32) {
        if (kv < cikk) val = w1[oc * cikk + kv];
    } else {
        const int kr = kv - K1_32;      // only reachable when K2_32 > 0 (w2 != null)
        if (kr < cikk) val = w2[oc * cikk + kr];
    }
    wpf[idx] = (_Float16)val;
}

// ---------------------------------------------------------------------------
// One span of 32-wide K chunks, all belonging to a single branch.
// B is gathered ONCE per chunk (table via wave-uniform scalar loads,
// unconditional data loads + cndmask), then reused by MTN back-to-back WMMAs
// with consecutive pre-packed A fragments.
// ---------------------------------------------------------------------------
template <int MTN>
__device__ __forceinline__ void gemm_chunks(
    const float* __restrict__ xb, const int2* __restrict__ tbl,
    const _Float16* __restrict__ wfrag,    // wpf + lane*16
    int kcBegin, int kcEnd, int tblBase, bool hiHalf,
    int base0, int base1, unsigned lim0, unsigned lim1, int laneBase,
    v8f acc[MTN])
{
    for (int kc = kcBegin; kc < kcEnd; ++kc) {
        const int tb = tblBase + (kc - kcBegin) * 32;
        v16h bfrag;
        #pragma unroll
        for (int tt = 0; tt < 16; ++tt) {
            const int2 e0 = tbl[tb + (tt < 8 ? tt : tt + 8)];        // wave-uniform -> s_load
            const int2 e1 = tbl[tb + 8 + (tt < 8 ? tt : tt + 8)];    // wave-uniform -> s_load
            const int off = hiHalf ? e1.x : e0.x;
            const int dpk = hiHalf ? e1.y : e0.y;
            const int a0 = base0 + (int)((unsigned)dpk >> 16);
            const int a1 = base1 + (dpk & 0xffff);
            const bool ok = ((unsigned)a0 < lim0) & ((unsigned)a1 < lim1);
            const int idx = ok ? (laneBase + off) : 0;
            const float v = xb[idx];             // unconditional, idx 0 is safe
            bfrag[tt] = (_Float16)(ok ? v : 0.0f);
        }
        #pragma unroll
        for (int mt = 0; mt < MTN; ++mt) {
            const v16h afrag = *(const v16h*)(wfrag + (kc * MTN + mt) * 512);
            acc[mt] = __builtin_amdgcn_wmma_f32_16x16x32_f16(
                          false, afrag, false, bfrag, (short)0, acc[mt],
                          false, false);
        }
    }
}

// ---------------------------------------------------------------------------
// Fused CenterPivotConv4d (both branches) as implicit GEMM on v_wmma f16->f32.
// One wave owns a (MTN*16) x 16 output tile; 8 waves per block.
// ---------------------------------------------------------------------------
template <int MTN>
__global__ __launch_bounds__(256) void cpc4d_wmma(
    const float* __restrict__ x, float* __restrict__ y,
    const _Float16* __restrict__ wpf, const int2* __restrict__ tbl,
    const float* __restrict__ bs,
    int ci, int ha, int wa, int hb, int wb,
    int s, int hbp, int wbp, int KC1, int KC, int K1_32,
    int tiles, int do_relu)
{
    const int lane = threadIdx.x & 31;
    const int wid  = blockIdx.x * (blockDim.x >> 5) + (threadIdx.x >> 5);
    if (wid >= tiles) return;            // wave-uniform: EXEC all-ones at WMMA

    const int co   = MTN * 16;
    const int Nsp  = ha * wa * hbp * wbp;
    const int nt_n = (Nsp + 15) >> 4;
    const int nt = wid % nt_n;
    const int bi = wid / nt_n;

    // decode this lane's N column -> (qi,qj,sp,sq)
    const int n = nt * 16 + (lane & 15);
    const bool nvalid = (n < Nsp);
    int qi = 0, qj = 0, sp = 0, sq = 0;
    {
        int r = nvalid ? n : 0;
        sq = r % wbp; r /= wbp;
        sp = r % hbp; r /= hbp;
        qj = r % wa;  qi = r / wa;
    }
    const int st_j = hb * wb, st_i = wa * st_j, st_c = ha * st_i;
    const int spb = sp * s, sqb = sq * s;
    const int laneBase = qi * st_i + qj * st_j + spb * wb + sqb;
    const float* __restrict__ xb = x + (long)bi * ci * st_c;

    // bounds bases (bias 256 folded in); poison when this column is padding
    int bq0 = qi - 256, bq1 = qj - 256, bs0 = spb - 256, bs1 = sqb - 256;
    if (!nvalid) { bq0 = bq1 = bs0 = bs1 = 0x20000000; }

    const bool hiHalf = (lane >= 16);
    const _Float16* __restrict__ wfrag = wpf + lane * 16;

    v8f acc[MTN] = {};
    // branch 1: conv over query dims at support pivot
    gemm_chunks<MTN>(xb, tbl, wfrag, 0, KC1, 0, hiHalf,
                     bq0, bq1, (unsigned)ha, (unsigned)wa, laneBase, acc);
    // branch 2: strided conv over support dims at query position
    gemm_chunks<MTN>(xb, tbl, wfrag, KC1, KC, K1_32, hiHalf,
                     bs0, bs1, (unsigned)hb, (unsigned)wb, laneBase, acc);

    // epilogue: bias (+ optional ReLU), store with 32-bit offsets
    const int ncol   = nt * 16 + (lane & 15);
    const int rowoff = hiHalf ? 8 : 0;
    if (ncol < Nsp) {
        #pragma unroll
        for (int mt = 0; mt < MTN; ++mt) {
            const int ybase = (bi * co + mt * 16 + rowoff) * Nsp + ncol;
            #pragma unroll
            for (int e2 = 0; e2 < 8; ++e2) {
                float v = acc[mt][e2] + bs[mt * 16 + e2 + rowoff];
                if (do_relu) v = fmaxf(v, 0.0f);
                y[ybase + e2 * Nsp] = v;
            }
        }
    }
}

// ---------------------------------------------------------------------------
// GroupNorm (groups=4) — deterministic two-stage reduction + fused affine/ReLU
// ---------------------------------------------------------------------------
__global__ __launch_bounds__(256) void gn_partial(const float* __restrict__ y,
                                                  float* __restrict__ red,
                                                  long cnt, int nblk)
{
    const int bg = blockIdx.y;
    const float* base = y + (long)bg * cnt;
    float s = 0.0f, s2 = 0.0f;
    for (long idx = (long)blockIdx.x * blockDim.x + threadIdx.x; idx < cnt;
         idx += (long)nblk * blockDim.x) {
        const float v = base[idx];
        s += v; s2 += v * v;
    }
    __shared__ float sh[256], sh2[256];
    sh[threadIdx.x] = s; sh2[threadIdx.x] = s2;
    __syncthreads();
    for (int off = 128; off; off >>= 1) {
        if ((int)threadIdx.x < off) {
            sh[threadIdx.x]  += sh[threadIdx.x + off];
            sh2[threadIdx.x] += sh2[threadIdx.x + off];
        }
        __syncthreads();
    }
    if (threadIdx.x == 0) {
        red[((long)bg * nblk + blockIdx.x) * 2]     = sh[0];
        red[((long)bg * nblk + blockIdx.x) * 2 + 1] = sh2[0];
    }
}

__global__ void gn_finish(const float* __restrict__ red, float* __restrict__ stat,
                          long cnt, int nblk)
{
    __shared__ float sh[64], sh2[64];
    const int bg = blockIdx.x;
    float s = 0.0f, s2 = 0.0f;
    for (int i = threadIdx.x; i < nblk; i += blockDim.x) {
        s  += red[((long)bg * nblk + i) * 2];
        s2 += red[((long)bg * nblk + i) * 2 + 1];
    }
    sh[threadIdx.x] = s; sh2[threadIdx.x] = s2;
    __syncthreads();
    for (int off = 32; off; off >>= 1) {
        if ((int)threadIdx.x < off) {
            sh[threadIdx.x]  += sh[threadIdx.x + off];
            sh2[threadIdx.x] += sh2[threadIdx.x + off];
        }
        __syncthreads();
    }
    if (threadIdx.x == 0) {
        const float m   = sh[0] / (float)cnt;
        const float var = sh2[0] / (float)cnt - m * m;
        stat[bg * 2]     = m;
        stat[bg * 2 + 1] = rsqrtf(var + 1e-5f);
    }
}

__global__ void gn_apply(float* __restrict__ y, const float* __restrict__ stat,
                         const float* __restrict__ gw, const float* __restrict__ gb,
                         int co, long Nsp, int G, long total)
{
    const long idx = (long)blockIdx.x * blockDim.x + threadIdx.x;
    if (idx >= total) return;
    const int c = (int)((idx / Nsp) % co);
    const int b = (int)(idx / (Nsp * co));
    const int g = c / (co / G);
    const float m = stat[(b * G + g) * 2];
    const float r = stat[(b * G + g) * 2 + 1];
    const float v = (y[idx] - m) * r * gw[c] + gb[c];
    y[idx] = fmaxf(v, 0.0f);
}

// ---------------------------------------------------------------------------
// Bilinear (align_corners=True) resize over query dims of (b,c,hi,wi,sup),
// optionally fused with elementwise add. Also used for final 2x upsample.
// ---------------------------------------------------------------------------
__global__ void bilerp_add(const float* __restrict__ x, const float* __restrict__ add,
                           float* __restrict__ out,
                           int B, int C, int hi, int wi, int ho, int wo, int sup)
{
    const long total = (long)B * C * ho * wo * sup;
    const long idx = (long)blockIdx.x * blockDim.x + threadIdx.x;
    if (idx >= total) return;
    long r = idx;
    const int sp = (int)(r % sup); r /= sup;
    const int J  = (int)(r % wo);  r /= wo;
    const int I  = (int)(r % ho);  r /= ho;
    const int c  = (int)(r % C);
    const int b  = (int)(r / C);

    const float fy = (ho > 1) ? (float)I * (float)(hi - 1) / (float)(ho - 1) : 0.0f;
    const float fx = (wo > 1) ? (float)J * (float)(wi - 1) / (float)(wo - 1) : 0.0f;
    int i0 = (int)floorf(fy); i0 = max(0, min(i0, hi - 1));
    int j0 = (int)floorf(fx); j0 = max(0, min(j0, wi - 1));
    const int i1 = min(i0 + 1, hi - 1);
    const int j1 = min(j0 + 1, wi - 1);
    const float wy = fy - (float)i0;
    const float wx = fx - (float)j0;

    const float* xb = x + ((long)b * C + c) * hi * wi * sup;
    const float v00 = xb[((long)i0 * wi + j0) * sup + sp];
    const float v01 = xb[((long)i0 * wi + j1) * sup + sp];
    const float v10 = xb[((long)i1 * wi + j0) * sup + sp];
    const float v11 = xb[((long)i1 * wi + j1) * sup + sp];
    float v = (1.0f - wy) * ((1.0f - wx) * v00 + wx * v01)
            +        wy  * ((1.0f - wx) * v10 + wx * v11);
    if (add) v += add[idx];
    out[idx] = v;
}

__global__ void mean_sup(const float* __restrict__ x, float* __restrict__ out,
                         long n, int sup)
{
    const long idx = (long)blockIdx.x * blockDim.x + threadIdx.x;
    if (idx >= n) return;
    const float* p = x + idx * sup;
    float s = 0.0f;
    for (int i = 0; i < sup; ++i) s += p[i];
    out[idx] = s / (float)sup;
}

// ---------------------------------------------------------------------------
// Host orchestration
// ---------------------------------------------------------------------------
struct LayerP { const float *w1, *b1, *w2, *b2, *gw, *gb; };
struct Dim    { int ci, co, ha, wa, hb, wb, k, s; };

extern "C" void kernel_launch(void* const* d_in, const int* in_sizes, int n_in,
                              void* d_out, int out_size, void* d_ws, size_t ws_size,
                              hipStream_t stream)
{
    (void)n_in; (void)out_size; (void)ws_size;
    const float *corr4, *corr3, *corr2, *dw1, *db1, *dw2, *db2;
    LayerP L[15];

    // canonical layer order: 0-2 enc4, 3-5 enc3, 6-8 enc2, 9-11 enc43, 12-14 enc32
    const bool sorted = (in_sizes[0] == 50000000);   // corr2 first => jax-sorted keys
    if (!sorted) {
        corr4 = (const float*)d_in[0];
        corr3 = (const float*)d_in[1];
        corr2 = (const float*)d_in[2];
        int idx = 3;
        for (int l = 0; l < 15; ++l) {
            L[l].w1 = (const float*)d_in[idx++]; L[l].b1 = (const float*)d_in[idx++];
            L[l].w2 = (const float*)d_in[idx++]; L[l].b2 = (const float*)d_in[idx++];
            L[l].gw = (const float*)d_in[idx++]; L[l].gb = (const float*)d_in[idx++];
        }
        dw1 = (const float*)d_in[idx++]; db1 = (const float*)d_in[idx++];
        dw2 = (const float*)d_in[idx++]; db2 = (const float*)d_in[idx++];
    } else {
        corr2 = (const float*)d_in[0];
        corr3 = (const float*)d_in[1];
        corr4 = (const float*)d_in[2];
        db1 = (const float*)d_in[3]; db2 = (const float*)d_in[4];
        dw1 = (const float*)d_in[5]; dw2 = (const float*)d_in[6];
        int idx = 7;
        const int order[5] = {6, 3, 12, 0, 9};       // enc2, enc3, enc32, enc4, enc43
        for (int bl = 0; bl < 5; ++bl)
            for (int li = 0; li < 3; ++li) {
                LayerP& P = L[order[bl] + li];
                P.b1 = (const float*)d_in[idx++]; P.b2 = (const float*)d_in[idx++];
                P.gb = (const float*)d_in[idx++]; P.gw = (const float*)d_in[idx++];
                P.w1 = (const float*)d_in[idx++]; P.w2 = (const float*)d_in[idx++];
            }
    }

    // workspace layout (floats)
    float* ws = (float*)d_ws;
    size_t o = 0;
    float* SQ4  = ws + o; o += 173056;     // (2,128,13,13,2,2)
    float* SQ3  = ws + o; o += 640000;     // (2,128,25,25,2,2)
    float* SQ2  = ws + o; o += 2560000;    // (2,128,50,50,2,2)
    float* RED  = ws + o; o += 1024;       // 8 groups * 64 blocks * 2
    float* STAT = ws + o; o += 16;
    float* BS   = ws + o; o += 128;
    _Float16* WPF = (_Float16*)(ws + o); o += 150000;   // up to 2304*128 halves
    int2* TBL = (int2*)(ws + o); o += 4608;             // up to 2304 int2
    float* PA   = ws + o; o += 27040000;   // enc2 L0 output
    float* PB   = ws + o; o += 5120000;

    const Dim D[15] = {
        {3,  32, 13, 13, 13, 13, 3, 2}, {32, 64, 13, 13, 7, 7, 3, 2}, {64, 128, 13, 13, 4, 4, 3, 2},
        {6,  32, 25, 25, 25, 25, 5, 4}, {32, 64, 25, 25, 7, 7, 3, 2}, {64, 128, 25, 25, 4, 4, 3, 2},
        {4,  32, 50, 50, 50, 50, 5, 4}, {32, 64, 50, 50, 13, 13, 5, 4}, {64, 128, 50, 50, 4, 4, 3, 2},
        {128, 128, 25, 25, 2, 2, 3, 1}, {128, 128, 25, 25, 2, 2, 3, 1}, {128, 128, 25, 25, 2, 2, 3, 1},
        {128, 128, 50, 50, 2, 2, 3, 1}, {128, 128, 50, 50, 2, 2, 3, 1}, {128, 128, 50, 50, 2, 2, 3, 1}};

    auto conv = [&](const float* xin, float* yout,
                    const float* w1, const float* b1, const float* w2, const float* b2,
                    Dim d, int relu) {
        const int hbp = (d.hb + d.s - 1) / d.s, wbp = (d.wb + d.s - 1) / d.s;
        const int cikk = d.ci * d.k * d.k;
        const int K1_32 = ((cikk + 31) / 32) * 32;
        const int K2_32 = w2 ? K1_32 : 0;
        const int K32 = K1_32 + K2_32;
        const int st_j = d.hb * d.wb, st_i = d.wa * st_j, st_c = d.ha * st_i;
        const int total_wpf = K32 * d.co;
        pack_wb<<<(total_wpf + 255) / 256, 256, 0, stream>>>(
            w1, b1, w2, b2, WPF, TBL, BS, d.co, cikk, K1_32, K2_32, d.k,
            st_c, st_i, st_j, d.wb, total_wpf);
        const long Nsp = (long)d.ha * d.wa * hbp * wbp;
        const int tiles = (int)(2L * ((Nsp + 15) / 16));
        const int blocks = (tiles + 7) / 8;
        const int KC1 = K1_32 / 32, KC = K32 / 32;
        switch (d.co / 16) {
        case 2:
            cpc4d_wmma<2><<<blocks, 256, 0, stream>>>(xin, yout, WPF, TBL, BS,
                d.ci, d.ha, d.wa, d.hb, d.wb, d.s, hbp, wbp, KC1, KC, K1_32,
                tiles, relu);
            break;
        case 4:
            cpc4d_wmma<4><<<blocks, 256, 0, stream>>>(xin, yout, WPF, TBL, BS,
                d.ci, d.ha, d.wa, d.hb, d.wb, d.s, hbp, wbp, KC1, KC, K1_32,
                tiles, relu);
            break;
        default:
            cpc4d_wmma<8><<<blocks, 256, 0, stream>>>(xin, yout, WPF, TBL, BS,
                d.ci, d.ha, d.wa, d.hb, d.wb, d.s, hbp, wbp, KC1, KC, K1_32,
                tiles, relu);
            break;
        }
    };

    auto gn = [&](float* y, int li, Dim d) {
        const int hbp = (d.hb + d.s - 1) / d.s, wbp = (d.wb + d.s - 1) / d.s;
        const long Nsp = (long)d.ha * d.wa * hbp * wbp;
        const long cnt = (long)(d.co / 4) * Nsp;
        dim3 g1(64, 8);
        gn_partial<<<g1, 256, 0, stream>>>(y, RED, cnt, 64);
        gn_finish<<<8, 64, 0, stream>>>(RED, STAT, cnt, 64);
        const long total = 2L * d.co * Nsp;
        gn_apply<<<(unsigned)((total + 255) / 256), 256, 0, stream>>>(
            y, STAT, L[li].gw, L[li].gb, d.co, Nsp, 4, total);
    };

    auto enc_layer = [&](const float* xin, float* yout, int li) {
        conv(xin, yout, L[li].w1, L[li].b1, L[li].w2, L[li].b2, D[li], 0);
        gn(yout, li, D[li]);
    };

    // enc4: corr4 -> SQ4
    enc_layer(corr4, PA, 0); enc_layer(PA, PB, 1); enc_layer(PB, SQ4, 2);
    // enc3: corr3 -> SQ3
    enc_layer(corr3, PA, 3); enc_layer(PA, PB, 4); enc_layer(PB, SQ3, 5);
    // enc2: corr2 -> SQ2
    enc_layer(corr2, PA, 6); enc_layer(PA, PB, 7); enc_layer(PB, SQ2, 8);

    // merge 4 -> 3: interp SQ4 (13->25) + SQ3, then enc43
    {
        const long total = 2L * 128 * 25 * 25 * 4;
        bilerp_add<<<(unsigned)((total + 255) / 256), 256, 0, stream>>>(
            SQ4, SQ3, PA, 2, 128, 13, 13, 25, 25, 4);
    }
    enc_layer(PA, PB, 9); enc_layer(PB, PA, 10); enc_layer(PA, PB, 11);

    // merge 3 -> 2: interp (25->50) + SQ2, then enc32
    {
        const long total = 2L * 128 * 50 * 50 * 4;
        bilerp_add<<<(unsigned)((total + 255) / 256), 256, 0, stream>>>(
            PB, SQ2, PA, 2, 128, 25, 25, 50, 50, 4);
    }
    enc_layer(PA, PB, 12); enc_layer(PB, PA, 13); enc_layer(PA, PB, 14);

    // mean over support positions: (2,128,50,50,4) -> (2,128,50,50)
    {
        const long n = 2L * 128 * 50 * 50;
        mean_sup<<<(unsigned)((n + 255) / 256), 256, 0, stream>>>(PB, PA, n, 4);
    }

    // decoder: two 3x3 convs with ReLU (reuse WMMA conv, hb=wb=1, no branch 2)
    const Dim dd1 = {128, 128, 50, 50, 1, 1, 3, 1};
    const Dim dd2 = {128, 64,  50, 50, 1, 1, 3, 1};
    conv(PA, PB, dw1, db1, nullptr, nullptr, dd1, 1);
    conv(PB, PA, dw2, db2, nullptr, nullptr, dd2, 1);

    // final x2 bilinear upsample (align_corners) -> d_out (2,64,100,100)
    {
        const long total = 2L * 64 * 100 * 100;
        bilerp_add<<<(unsigned)((total + 255) / 256), 256, 0, stream>>>(
            PA, nullptr, (float*)d_out, 2, 64, 50, 50, 100, 100, 1);
    }
}
